// BenchmarkRBM_56289841381582
// MI455X (gfx1250) — compile-verified, matrix operand order checked by reading, not run on hardware
//
#include <hip/hip_runtime.h>
#include <cstdint>
#include <cstddef>

// Problem constants (match reference)
#define BDIM 16384
#define VDIM 1024
#define HDIM 1024
#define CDIM 64
#define KSTEPS 25

typedef __attribute__((ext_vector_type(16))) __bf16        v16bf;
typedef __attribute__((ext_vector_type(8)))  float         v8f;
typedef __attribute__((ext_vector_type(4)))  unsigned int  v4u;

union V16 { v16bf v; v4u q[2]; };

__device__ __forceinline__ unsigned short f2bf(float f) {
  unsigned u = __builtin_bit_cast(unsigned, f);
  u += 0x7FFFu + ((u >> 16) & 1u);          // round-to-nearest-even
  return (unsigned short)(u >> 16);
}
__device__ __forceinline__ float bf2f(unsigned short h) {
  return __builtin_bit_cast(float, (unsigned)h << 16);
}
__device__ __forceinline__ uint64_t mix64(uint64_t x) {
  x += 0x9E3779B97F4A7C15ull;
  x = (x ^ (x >> 30)) * 0xBF58476D1CE4E5B9ull;
  x = (x ^ (x >> 27)) * 0x94D049BB133111EBull;
  return x ^ (x >> 31);
}
__device__ __forceinline__ float u01(uint64_t x) {
  return (float)(mix64(x) >> 40) * (1.0f / 16777216.0f);
}

// ---------------------------------------------------------------------------
// CDNA5 async global->LDS copy (16B), tracked by ASYNCcnt.
// LDS address is taken from the vdst VGPR; generic LDS pointer's low 32 bits are the
// LDS byte offset (ISA 10.2 aperture mapping).
// ---------------------------------------------------------------------------
__device__ __forceinline__ void async_cp16(const unsigned short* g, unsigned short* l) {
  unsigned int loff = (unsigned int)(unsigned long long)(uintptr_t)l;
  asm volatile("global_load_async_to_lds_b128 %0, %1, off"
               : : "v"(loff), "v"(g) : "memory");
}
__device__ __forceinline__ void async_wait_le4() {
  asm volatile("s_wait_asynccnt 0x4" ::: "memory");
}
__device__ __forceinline__ void async_wait_0() {
  asm volatile("s_wait_asynccnt 0x0" ::: "memory");
}

// ---------------------------------------------------------------------------
// Tiny helper kernels
// ---------------------------------------------------------------------------
__global__ void zero_out_k(float* o) {
  if (threadIdx.x == 0 && blockIdx.x == 0) o[0] = 0.0f;
}

// W (V,H) f32 -> Wbf (V,H) bf16 row-major  and  WTbf (H,V) bf16 row-major
__global__ __launch_bounds__(256) void convert_W_k(const float* __restrict__ W,
                                                   unsigned short* __restrict__ Wbf,
                                                   unsigned short* __restrict__ WTbf) {
  int idx = blockIdx.x * blockDim.x + threadIdx.x;  // V*H
  int v = idx >> 10, h = idx & 1023;
  unsigned short x = f2bf(W[idx]);
  Wbf[idx] = x;
  WTbf[(size_t)h * VDIM + v] = x;
}

__global__ __launch_bounds__(256) void convert_f32_bf16_k(const float* __restrict__ in,
                                                          unsigned short* __restrict__ out) {
  size_t i = (size_t)blockIdx.x * blockDim.x + threadIdx.x;
  out[i] = f2bf(in[i]);
}

// v_start ~ Bernoulli(0.5), deterministic counter RNG
__global__ __launch_bounds__(256) void init_v_k(unsigned short* __restrict__ v, uint64_t seed) {
  size_t i = (size_t)blockIdx.x * blockDim.x + threadIdx.x;
  v[i] = (mix64(seed + i) & 1ull) ? (unsigned short)0x3F80 : (unsigned short)0;
}

// t = tanh(cond @ W1 + b1), one thread per (b,j), j in [0,64)
__global__ __launch_bounds__(256) void param_t_k(const float* __restrict__ cond,
                                                 const float* __restrict__ W1,
                                                 const float* __restrict__ b1,
                                                 float* __restrict__ t) {
  int idx = blockIdx.x * blockDim.x + threadIdx.x;  // B*64
  int b = idx >> 6, j = idx & 63;
  const float* cr = cond + (size_t)b * CDIM;
  float s = b1[j];
#pragma unroll 8
  for (int k = 0; k < CDIM; ++k) s += cr[k] * W1[k * 64 + j];
  t[idx] = tanhf(s);
}

// b_mod/c_mod: one thread per (b,j), j in [0,1024). 4 dots of length 64.
__global__ __launch_bounds__(256) void param_mod_k(const float* __restrict__ t,
                                                   const float* __restrict__ W2,
                                                   const float* __restrict__ b2,
                                                   const float* __restrict__ bvec,
                                                   const float* __restrict__ cvec,
                                                   float* __restrict__ bmod,
                                                   float* __restrict__ cmod) {
  int idx = blockIdx.x * blockDim.x + threadIdx.x;  // B*1024
  int b = idx >> 10, j = idx & 1023;
  const float* tr = t + (size_t)b * 64;
  float gb = b2[j], bb = b2[1024 + j], gc = b2[2048 + j], bc = b2[3072 + j];
#pragma unroll 4
  for (int k = 0; k < 64; ++k) {
    float tk = tr[k];
    const float* w = W2 + (size_t)k * 4096 + j;
    gb += tk * w[0];
    bb += tk * w[1024];
    gc += tk * w[2048];
    bc += tk * w[3072];
  }
  bmod[idx] = (1.0f + gb) * bvec[j] + bb;
  cmod[idx] = (1.0f + gc) * cvec[j] + bc;
}

// loss += sum((v_model - v_data) * b_mod) / B
__global__ __launch_bounds__(256) void dot_term_k(const float* __restrict__ vdat,
                                                  const unsigned short* __restrict__ vmod,
                                                  const float* __restrict__ bmod,
                                                  float* __restrict__ loss) {
  size_t n = (size_t)BDIM * VDIM;
  float local = 0.0f;
  for (size_t i = (size_t)blockIdx.x * blockDim.x + threadIdx.x; i < n;
       i += (size_t)gridDim.x * blockDim.x)
    local += (bf2f(vmod[i]) - vdat[i]) * bmod[i];
  atomicAdd(loss, local * (1.0f / (float)BDIM));
}

// ---------------------------------------------------------------------------
// Fused WMMA GEMM: pre = X(B,K) @ WR(N,K)^T + bias(B,N)
//   MODE 0: sample  -> OUT[b,n] = bernoulli(sigmoid(pre)) in bf16
//   MODE 1: free en -> atomicAdd(loss, scale * sum softplus(pre))
//
// Block: 256 threads (8 waves), output tile 128(M) x 128(N).
// Wave (wm in 0..3, wn in 0..1) owns 32x64 = 2x4 WMMA tiles.
// K-slices of 32 are double-buffered in LDS; staging uses CDNA5 async
// global->LDS b128 copies (ASYNCcnt), so stage s+1 streams in while stage s
// feeds the matrix cores.
// 16-bit operand layout (ISA 7.12.2): per lane, two contiguous 16B chunks of
// the operand row at k-offsets {8*(lane>>4), 8*(lane>>4)+16}.
// ---------------------------------------------------------------------------
#define LDS_STRIDE 40  // halves per row (80B, 16B-aligned, bank-spreading pad)

template <int MODE>
__global__ __launch_bounds__(256) void rbm_gemm_k(const unsigned short* __restrict__ X,
                                                  const unsigned short* __restrict__ WR,
                                                  const float* __restrict__ bias,
                                                  unsigned short* __restrict__ OUT,
                                                  int K, int N, uint64_t seed,
                                                  float* __restrict__ loss, float scale) {
  __shared__ __align__(16) unsigned short sX[2][128][LDS_STRIDE];
  __shared__ __align__(16) unsigned short sW[2][128][LDS_STRIDE];

  const int tid  = threadIdx.x;
  const int lane = tid & 31;
  const int wave = tid >> 5;
  const int wm   = wave & 3;   // 0..3 -> 32-row slab
  const int wn   = wave >> 2;  // 0..1 -> 64-col slab
  const int bm   = blockIdx.y * 128;
  const int bn   = blockIdx.x * 128;

  const int rlo   = lane & 15;
  const int khalf = (lane >> 4) << 3;  // 0 or 8
  const int hi8   = (lane >> 4) << 3;  // C/D row offset for lanes 16..31

  v8f acc[2][4];
#pragma unroll
  for (int tm = 0; tm < 2; ++tm)
#pragma unroll
    for (int tn = 0; tn < 4; ++tn) acc[tm][tn] = (v8f){};

  // Issue one K-slice (128x32 X-tile + 128x32 W-tile) as 4 async b128 ops
  // per thread (1024 chunks / 256 threads), fully uniform (no EXEC games).
  auto issue_stage = [&](int kk, int buf) {
#pragma unroll
    for (int i = 0; i < 2; ++i) {
      int c = tid + i * 256;               // 0..511
      int row = c >> 2, cp = (c & 3) << 3; // cp in {0,8,16,24}
      async_cp16(X + (size_t)(bm + row) * K + kk + cp, &sX[buf][row][cp]);
    }
#pragma unroll
    for (int i = 0; i < 2; ++i) {
      int c = tid + i * 256;
      int row = c >> 2, cp = (c & 3) << 3;
      async_cp16(WR + (size_t)(bn + row) * K + kk + cp, &sW[buf][row][cp]);
    }
  };

  auto compute = [&](int buf) {
    const unsigned short (*Xb)[LDS_STRIDE] = sX[buf];
    const unsigned short (*Wb)[LDS_STRIDE] = sW[buf];
    V16 a[2], b[4];
#pragma unroll
    for (int tm = 0; tm < 2; ++tm) {
      int r = wm * 32 + tm * 16 + rlo;
      a[tm].q[0] = *(const v4u*)&Xb[r][khalf];
      a[tm].q[1] = *(const v4u*)&Xb[r][khalf + 16];
    }
#pragma unroll
    for (int tn = 0; tn < 4; ++tn) {
      int r = wn * 64 + tn * 16 + rlo;
      b[tn].q[0] = *(const v4u*)&Wb[r][khalf];
      b[tn].q[1] = *(const v4u*)&Wb[r][khalf + 16];
    }
#pragma unroll
    for (int tm = 0; tm < 2; ++tm)
#pragma unroll
      for (int tn = 0; tn < 4; ++tn)
        acc[tm][tn] = __builtin_amdgcn_wmma_f32_16x16x32_bf16(
            false, a[tm].v, false, b[tn].v, (short)0, acc[tm][tn], false, false);
  };

  const int nstage = K >> 5;  // K/32
  issue_stage(0, 0);
  for (int s = 0; s < nstage; ++s) {
    const int buf = s & 1;
    if (s + 1 < nstage) {
      __syncthreads();                 // all waves done with buf^1 (stage s-1)
      issue_stage((s + 1) << 5, buf ^ 1);
      async_wait_le4();                // stage s done; stage s+1 in flight
    } else {
      async_wait_0();
    }
    __syncthreads();                   // stage-s data visible to all waves
    compute(buf);
  }

  float fe_local = 0.0f;

  // Epilogue per 16x16 tile. C/D layout: VGPR i holds M = i (+8 for hi lanes),
  // N = lane&15.
#pragma unroll
  for (int tm = 0; tm < 2; ++tm) {
#pragma unroll
    for (int tn = 0; tn < 4; ++tn) {
      const int r0 = bm + wm * 32 + tm * 16 + hi8;
      const int c0 = bn + wn * 64 + tn * 16 + rlo;
#pragma unroll
      for (int i = 0; i < 8; ++i) {
        const size_t idx = (size_t)(r0 + i) * N + c0;
        float z = acc[tm][tn][i] + bias[idx];  // T == 1.0
        if (MODE == 0) {
          float p = 1.0f / (1.0f + __expf(-z));
          float u = u01(seed + idx);
          OUT[idx] = (u < p) ? (unsigned short)0x3F80 : (unsigned short)0;
        } else {
          fe_local += (z > 20.0f) ? z : log1pf(__expf(z));
        }
      }
    }
  }

  if (MODE == 1) atomicAdd(loss, fe_local * scale);
}

// ---------------------------------------------------------------------------
// Host launch
// ---------------------------------------------------------------------------
extern "C" void kernel_launch(void* const* d_in, const int* in_sizes, int n_in,
                              void* d_out, int out_size, void* d_ws, size_t ws_size,
                              hipStream_t stream) {
  (void)in_sizes; (void)n_in; (void)out_size; (void)ws_size;

  const float* v_data = (const float*)d_in[0];  // (B,V)
  const float* cond   = (const float*)d_in[1];  // (B,C)
  const float* W      = (const float*)d_in[2];  // (V,H)
  const float* bvec   = (const float*)d_in[3];  // (V,)
  const float* cvec   = (const float*)d_in[4];  // (H,)
  const float* W1     = (const float*)d_in[5];  // (C,64)
  const float* b1     = (const float*)d_in[6];  // (64,)
  const float* W2     = (const float*)d_in[7];  // (64,4096)
  const float* b2     = (const float*)d_in[8];  // (4096,)
  float* loss         = (float*)d_out;          // scalar

  // Workspace carve-up (bytes)
  char* ws = (char*)d_ws;
  float*          b_mod    = (float*)(ws);                                       // 64 MiB
  float*          c_mod    = (float*)(ws + (size_t)64 * 1024 * 1024);            // 64 MiB
  unsigned short* vbuf     = (unsigned short*)(ws + (size_t)128 * 1024 * 1024);  // 32 MiB
  unsigned short* hbuf     = (unsigned short*)(ws + (size_t)160 * 1024 * 1024);  // 32 MiB
  unsigned short* vdata_bf = (unsigned short*)(ws + (size_t)192 * 1024 * 1024);  // 32 MiB
  unsigned short* Wbf      = (unsigned short*)(ws + (size_t)224 * 1024 * 1024);  // 2 MiB
  unsigned short* WTbf     = (unsigned short*)(ws + (size_t)226 * 1024 * 1024);  // 2 MiB
  float*          tbuf     = (float*)(ws + (size_t)228 * 1024 * 1024);           // 4 MiB

  zero_out_k<<<1, 64, 0, stream>>>(loss);

  convert_W_k<<<(VDIM * HDIM) / 256, 256, 0, stream>>>(W, Wbf, WTbf);
  convert_f32_bf16_k<<<(BDIM * VDIM) / 256, 256, 0, stream>>>(v_data, vdata_bf);

  param_t_k<<<(BDIM * CDIM) / 256, 256, 0, stream>>>(cond, W1, b1, tbuf);
  param_mod_k<<<(BDIM * 1024) / 256, 256, 0, stream>>>(tbuf, W2, b2, bvec, cvec, b_mod, c_mod);

  init_v_k<<<(BDIM * VDIM) / 256, 256, 0, stream>>>(vbuf, 0xC0FFEEull << 32);

  // Gibbs chain: v -> h (v@W + c_mod), h -> v (h@W^T + b_mod)
  dim3 blk(256);
  dim3 grid_h(HDIM / 128, BDIM / 128);  // (8, 128)
  dim3 grid_v(VDIM / 128, BDIM / 128);
  for (int k = 0; k < KSTEPS; ++k) {
    uint64_t s_h = ((uint64_t)(2 * k + 1)) << 36;
    uint64_t s_v = ((uint64_t)(2 * k + 2)) << 36;
    rbm_gemm_k<0><<<grid_h, blk, 0, stream>>>(vbuf, WTbf, c_mod, hbuf, VDIM, HDIM, s_h, nullptr, 0.0f);
    rbm_gemm_k<0><<<grid_v, blk, 0, stream>>>(hbuf, Wbf, b_mod, vbuf, HDIM, VDIM, s_v, nullptr, 0.0f);
  }

  // loss += [-softplus(v_data@W + c_mod) + softplus(v_model@W + c_mod)] / B
  rbm_gemm_k<1><<<grid_h, blk, 0, stream>>>(vdata_bf, WTbf, c_mod, nullptr, VDIM, HDIM, 0, loss, -1.0f / (float)BDIM);
  rbm_gemm_k<1><<<grid_h, blk, 0, stream>>>(vbuf,     WTbf, c_mod, nullptr, VDIM, HDIM, 0, loss, +1.0f / (float)BDIM);

  // loss += sum((v_model - v_data) * b_mod) / B
  dot_term_k<<<2048, 256, 0, stream>>>(v_data, vbuf, b_mod, loss);
}
